// MotionFormerBlock_83631603188471
// MI455X (gfx1250) — matrix operand order repaired and checked by reading, not compiled, not run on hardware
//
#include <hip/hip_runtime.h>
#include <math.h>

typedef __bf16 bf16;
typedef bf16  v16bf __attribute__((ext_vector_type(16)));
typedef bf16  v8bf  __attribute__((ext_vector_type(8)));
typedef float v8f   __attribute__((ext_vector_type(8)));

// ---------------------------------------------------------------------------
// Problem constants (hardcoded from the reference)
// ---------------------------------------------------------------------------
static constexpr int DIMC   = 256;
static constexpr int MDIM   = 128;
static constexpr int HIDDEN = 1024;
static constexpr int NTOK   = 131072;     // B*H*W = 8*128*128
static constexpr int NWIN   = 2048;       // 8 * 16 * 16 windows

// ---------------------------------------------------------------------------
// WMMA helpers (CDNA5 gfx1250, wave32)
// ---------------------------------------------------------------------------
__device__ inline v8f v8f_zero() {
  v8f z;
#pragma unroll
  for (int i = 0; i < 8; ++i) z[i] = 0.0f;
  return z;
}

__device__ inline v8f wmma_bf16(v16bf a, v16bf b, v8f c) {
  return __builtin_amdgcn_wmma_f32_16x16x32_bf16(false, a, false, b, (short)0, c,
                                                 false, false);
}

// A-operand (16x32 bf16) per-lane fragment load from an LDS row-major tile.
// k order used here (B matches): lane m (0..15 / 16..31), VGPR v holds K = half*16+v*2, +1
__device__ inline v16bf load_a_frag(const bf16* base, int stride) {
  int lane = threadIdx.x & 31;
  int m = lane & 15;
  int half = (lane >> 4) & 1;
  const bf16* p = base + m * stride;
  v16bf a;
#pragma unroll
  for (int v = 0; v < 8; ++v) {
    int k2 = half * 16 + v * 2;
    a[2 * v]     = p[k2];
    a[2 * v + 1] = p[k2 + 1];
  }
  return a;
}

// B-operand (32x16 bf16). Tile held as [N][K] (row = N index, K contiguous):
// lane n%16 -> column N, half-wave selects K 0..15 / 16..31 (contiguous 32B).
__device__ inline v16bf load_b_frag(const bf16* base, int stride) {
  int lane = threadIdx.x & 31;
  int n = lane & 15;
  int koff = (lane >> 4) * 16;
  const v8bf* p = (const v8bf*)(base + n * stride + koff);
  v8bf lo = p[0];
  v8bf hi = p[1];
  v16bf b;
#pragma unroll
  for (int e = 0; e < 8; ++e) { b[e] = lo[e]; b[8 + e] = hi[e]; }
  return b;
}

// window-layout row -> flat (B, H*W) token index (window reverse)
__device__ inline size_t remap_row(int rw) {
  int win = rw >> 6, t = rw & 63;
  int b = win >> 8, wi = win & 255;
  int wh = wi >> 4, ww = wi & 15;
  int th = t >> 3, tw = t & 7;
  int h = wh * 8 + th, w = ww * 8 + tw;
  return (size_t)b * 16384 + (size_t)h * 128 + w;
}

// ---------------------------------------------------------------------------
// Weight prep: W (KxN f32, row major) -> WT (NxK bf16, row major)
// ---------------------------------------------------------------------------
__global__ __launch_bounds__(256) void mfb_transpose(const float* __restrict__ W,
                                                     bf16* __restrict__ WT,
                                                     int K, int N) {
  int i = blockIdx.x * 256 + threadIdx.x;
  if (i >= K * N) return;
  int n = i / K;
  int k = i - n * K;
  WT[i] = (bf16)W[(size_t)k * N + n];
}

// ---------------------------------------------------------------------------
// LN1 + coordinate embedding. One wave (32 lanes) per token, 8 tokens / block.
// Writes x_norm (f32) and ce (f32) in WINDOW layout.
// ---------------------------------------------------------------------------
__global__ __launch_bounds__(256) void mfb_ln1_ce(
    const float* __restrict__ x, const float* __restrict__ cor,
    const float* __restrict__ g1, const float* __restrict__ b1,
    const float* __restrict__ Wce, const float* __restrict__ bce,
    float* __restrict__ xn, float* __restrict__ cef) {
  int wave = threadIdx.x >> 5, lane = threadIdx.x & 31;
  int gtok = blockIdx.x * 8 + wave;
  const float* xr = x + (size_t)gtok * DIMC;
  int c0 = lane * 8;
  float v[8];
  float4 f0 = *(const float4*)(xr + c0);
  float4 f1 = *(const float4*)(xr + c0 + 4);
  v[0] = f0.x; v[1] = f0.y; v[2] = f0.z; v[3] = f0.w;
  v[4] = f1.x; v[5] = f1.y; v[6] = f1.z; v[7] = f1.w;
  float s = 0.f, ss = 0.f;
#pragma unroll
  for (int j = 0; j < 8; ++j) { s += v[j]; ss += v[j] * v[j]; }
#pragma unroll
  for (int off = 16; off >= 1; off >>= 1) {
    s  += __shfl_xor(s,  off, 32);
    ss += __shfl_xor(ss, off, 32);
  }
  float mean = s * (1.0f / 256.0f);
  float var  = ss * (1.0f / 256.0f) - mean * mean;
  float rstd = rsqrtf(var + 1e-5f);

  int b = gtok >> 14, rem = gtok & 16383;
  int h = rem >> 7, w = rem & 127;
  int win = (b << 8) + ((h >> 3) << 4) + (w >> 3);
  int tt = ((h & 7) << 3) + (w & 7);
  size_t rw = (size_t)win * 64 + tt;

  float* orow = xn + rw * DIMC;
#pragma unroll
  for (int j = 0; j < 8; ++j)
    orow[c0 + j] = (v[j] - mean) * rstd * g1[c0 + j] + b1[c0 + j];

  float ca = cor[(size_t)gtok * 2 + 0];
  float cb = cor[(size_t)gtok * 2 + 1];
  int j0 = lane * 4;
  float* crow = cef + rw * MDIM;
#pragma unroll
  for (int j = 0; j < 4; ++j)
    crow[j0 + j] = ca * Wce[j0 + j] + cb * Wce[128 + j0 + j] + bce[j0 + j];
}

// ---------------------------------------------------------------------------
// LN2 (flat layout, bf16 out)
// ---------------------------------------------------------------------------
__global__ __launch_bounds__(256) void mfb_ln2(
    const float* __restrict__ X, const float* __restrict__ g2,
    const float* __restrict__ b2, bf16* __restrict__ out) {
  int wave = threadIdx.x >> 5, lane = threadIdx.x & 31;
  int gtok = blockIdx.x * 8 + wave;
  const float* xr = X + (size_t)gtok * DIMC;
  int c0 = lane * 8;
  float v[8];
  float4 f0 = *(const float4*)(xr + c0);
  float4 f1 = *(const float4*)(xr + c0 + 4);
  v[0] = f0.x; v[1] = f0.y; v[2] = f0.z; v[3] = f0.w;
  v[4] = f1.x; v[5] = f1.y; v[6] = f1.z; v[7] = f1.w;
  float s = 0.f, ss = 0.f;
#pragma unroll
  for (int j = 0; j < 8; ++j) { s += v[j]; ss += v[j] * v[j]; }
#pragma unroll
  for (int off = 16; off >= 1; off >>= 1) {
    s  += __shfl_xor(s,  off, 32);
    ss += __shfl_xor(ss, off, 32);
  }
  float mean = s * (1.0f / 256.0f);
  float rstd = rsqrtf(ss * (1.0f / 256.0f) - mean * mean + 1e-5f);
  bf16* orow = out + (size_t)gtok * DIMC;
#pragma unroll
  for (int j = 0; j < 8; ++j)
    orow[c0 + j] = (bf16)((v[j] - mean) * rstd * g2[c0 + j] + b2[c0 + j]);
}

// ---------------------------------------------------------------------------
// Tiled WMMA GEMM: C[M,N] = A[M,K] @ BT[N,K]^T + bias (+resid) (epilogue opts)
// BM=BN=128, BK=32, 256 threads = 8 waves in 2x4, wave tile 64x32.
// ---------------------------------------------------------------------------
static constexpr int GAS = 40;  // LDS A row stride (elems)
static constexpr int GBS = 40;  // LDS B row stride (elems)

__device__ inline void stage_a_tile(const float* A, bf16* sA, int M0, int K,
                                    int k0, int t) {
#pragma unroll
  for (int i = 0; i < 4; ++i) {
    int e = i * 256 + t;
    int row = e >> 3;
    int col = (e & 7) * 4;
    float4 f = *(const float4*)(A + (size_t)(M0 + row) * K + k0 + col);
    bf16* d = sA + row * GAS + col;
    d[0] = (bf16)f.x; d[1] = (bf16)f.y; d[2] = (bf16)f.z; d[3] = (bf16)f.w;
  }
}

__device__ inline void stage_a_tile(const bf16* A, bf16* sA, int M0, int K,
                                    int k0, int t) {
#pragma unroll
  for (int i = 0; i < 2; ++i) {
    int e = i * 256 + t;
    int row = e >> 2;
    int col = (e & 3) * 8;
    *(v8bf*)(sA + row * GAS + col) =
        *(const v8bf*)(A + (size_t)(M0 + row) * K + k0 + col);
  }
}

template <typename AT>
__global__ __launch_bounds__(256) void mfb_gemm(
    const AT* __restrict__ A, const bf16* __restrict__ BT,
    const float* __restrict__ bias, const float* __restrict__ resid,
    float* __restrict__ outF, bf16* __restrict__ outB, int M, int N, int K,
    int remap) {
  __shared__ __align__(16) bf16 sA[128 * GAS];
  __shared__ __align__(16) bf16 sB[128 * GBS];
  int t = threadIdx.x;
  int wave = t >> 5, lane = t & 31;
  int wm = wave >> 2;   // 0..1
  int wn = wave & 3;    // 0..3
  int M0 = blockIdx.y * 128;
  int N0 = blockIdx.x * 128;

  v8f acc[4][2];
#pragma unroll
  for (int mt = 0; mt < 4; ++mt)
#pragma unroll
    for (int nt = 0; nt < 2; ++nt) acc[mt][nt] = v8f_zero();

  for (int k0 = 0; k0 < K; k0 += 32) {
    __syncthreads();
    stage_a_tile(A, sA, M0, K, k0, t);
#pragma unroll
    for (int i = 0; i < 2; ++i) {
      int e = i * 256 + t;
      int n = e >> 2;
      int col = (e & 3) * 8;
      *(v8bf*)(sB + n * GBS + col) =
          *(const v8bf*)(BT + (size_t)(N0 + n) * K + k0 + col);
    }
    if (k0 + 32 < K)  // hint next A tile into cache (global_prefetch_b8)
      __builtin_prefetch(A + (size_t)(M0 + (t >> 1)) * K + k0 + 32, 0, 1);
    __syncthreads();

    v16bf bfr[2];
#pragma unroll
    for (int nt = 0; nt < 2; ++nt)
      bfr[nt] = load_b_frag(sB + (wn * 32 + nt * 16) * GBS, GBS);
#pragma unroll
    for (int mt = 0; mt < 4; ++mt) {
      v16bf a = load_a_frag(sA + (wm * 64 + mt * 16) * GAS, GAS);
      acc[mt][0] = wmma_bf16(a, bfr[0], acc[mt][0]);
      acc[mt][1] = wmma_bf16(a, bfr[1], acc[mt][1]);
    }
  }

  int halfadd = (lane >= 16) ? 8 : 0;
  int nlane = lane & 15;
#pragma unroll
  for (int mt = 0; mt < 4; ++mt)
#pragma unroll
    for (int nt = 0; nt < 2; ++nt)
#pragma unroll
      for (int g = 0; g < 8; ++g) {
        int row = M0 + wm * 64 + mt * 16 + g + halfadd;
        int col = N0 + wn * 32 + nt * 16 + nlane;
        float v = acc[mt][nt][g] + bias[col];
        if (resid) v += resid[(size_t)row * N + col];
        if (outF) {
          size_t orow = remap ? remap_row(row) : (size_t)row;
          outF[orow * N + col] = v;
        } else {
          outB[(size_t)row * N + col] = (bf16)v;
        }
      }
}

// ---------------------------------------------------------------------------
// Windowed cross-frame attention. 1 block = 1 window (64 tokens),
// 8 waves = 8 heads. q from window r, k/v from paired window (r+1024)%2048,
// ce from window r. Produces af (attn@v, bf16) and (attn@ce - ce) (bf16).
// ---------------------------------------------------------------------------
static constexpr int AQS = 264;  // sQ / sK row stride
static constexpr int AVS = 72;   // sVT / sCT row stride
static constexpr int APS = 72;   // sP row stride

__global__ __launch_bounds__(256) void mfb_attn(
    const bf16* __restrict__ Q, const bf16* __restrict__ KV,
    const float* __restrict__ CE, bf16* __restrict__ AF,
    bf16* __restrict__ MF) {
  __shared__ __align__(16) bf16 sQ[64 * AQS];
  __shared__ __align__(16) bf16 sK[64 * AQS];
  __shared__ __align__(16) bf16 sVT[256 * AVS];   // [channel][token]
  __shared__ __align__(16) bf16 sCT[128 * AVS];   // [mchannel][token]
  __shared__ __align__(16) bf16 sP[8 * 16 * APS]; // per-head P row block

  int t = threadIdx.x;
  int r = blockIdx.x;
  int rp = (r + 1024) & 2047;
  size_t qbase = (size_t)r * 64 * DIMC;
  size_t kvbase = (size_t)rp * 64 * 512;
  size_t cbase = (size_t)r * 64 * MDIM;

#pragma unroll
  for (int i = 0; i < 8; ++i) {  // q rows
    int e = i * 256 + t;
    int row = e >> 5, col = (e & 31) * 8;
    *(v8bf*)(sQ + row * AQS + col) =
        *(const v8bf*)(Q + qbase + (size_t)row * DIMC + col);
  }
#pragma unroll
  for (int i = 0; i < 8; ++i) {  // k rows (cols 0..255 of kv)
    int e = i * 256 + t;
    int row = e >> 5, col = (e & 31) * 8;
    *(v8bf*)(sK + row * AQS + col) =
        *(const v8bf*)(KV + kvbase + (size_t)row * 512 + col);
  }
#pragma unroll
  for (int i = 0; i < 8; ++i) {  // v rows (cols 256..511), transpose into LDS
    int e = i * 256 + t;
    int tok = e >> 5, c8 = (e & 31) * 8;
    v8bf vv = *(const v8bf*)(KV + kvbase + (size_t)tok * 512 + 256 + c8);
#pragma unroll
    for (int j = 0; j < 8; ++j) sVT[(c8 + j) * AVS + tok] = vv[j];
  }
#pragma unroll
  for (int i = 0; i < 8; ++i) {  // ce rows (f32 -> bf16), transposed
    int e = i * 256 + t;
    int tok = e >> 5, c4 = (e & 31) * 4;
    float4 f = *(const float4*)(CE + cbase + (size_t)tok * MDIM + c4);
    sCT[(c4 + 0) * AVS + tok] = (bf16)f.x;
    sCT[(c4 + 1) * AVS + tok] = (bf16)f.y;
    sCT[(c4 + 2) * AVS + tok] = (bf16)f.z;
    sCT[(c4 + 3) * AVS + tok] = (bf16)f.w;
  }
  __syncthreads();

  int wave = t >> 5, lane = t & 31;
  int nlane = lane & 15;
  int halfadd = (lane >= 16) ? 8 : 0;
  int hh = wave;  // head
  bf16* sPh = sP + hh * 16 * APS;
  const float scale = 0.17677669529663687f;  // 1/sqrt(32)

  for (int mt = 0; mt < 4; ++mt) {
    // ---- S = q @ k^T : 16 query rows x 64 keys
    v16bf aQ = load_a_frag(sQ + (mt * 16) * AQS + hh * 32, AQS);
    v8f sv[4];
#pragma unroll
    for (int nt = 0; nt < 4; ++nt) {
      v16bf bK = load_b_frag(sK + (nt * 16) * AQS + hh * 32, AQS);
      sv[nt] = wmma_bf16(aQ, bK, v8f_zero());
    }
    // ---- row softmax in C layout: row m in lanes' halves, col = nt*16+nlane
#pragma unroll
    for (int g = 0; g < 8; ++g) {
      float m = fmaxf(fmaxf(sv[0][g], sv[1][g]), fmaxf(sv[2][g], sv[3][g]));
#pragma unroll
      for (int off = 8; off >= 1; off >>= 1) m = fmaxf(m, __shfl_xor(m, off, 32));
      float p0 = __expf((sv[0][g] - m) * scale);
      float p1 = __expf((sv[1][g] - m) * scale);
      float p2 = __expf((sv[2][g] - m) * scale);
      float p3 = __expf((sv[3][g] - m) * scale);
      float s = p0 + p1 + p2 + p3;
#pragma unroll
      for (int off = 8; off >= 1; off >>= 1) s += __shfl_xor(s, off, 32);
      float inv = 1.0f / s;
      int prow = (g + halfadd) * APS;
      sPh[prow + 0 * 16 + nlane] = (bf16)(p0 * inv);
      sPh[prow + 1 * 16 + nlane] = (bf16)(p1 * inv);
      sPh[prow + 2 * 16 + nlane] = (bf16)(p2 * inv);
      sPh[prow + 3 * 16 + nlane] = (bf16)(p3 * inv);
    }
    // (same-wave LDS RAW; hardware DScnt waits inserted by compiler)
    v16bf aP0 = load_a_frag(sPh + 0, APS);
    v16bf aP1 = load_a_frag(sPh + 32, APS);

    // ---- af = P @ v (N = 32 head channels)
#pragma unroll
    for (int nt2 = 0; nt2 < 2; ++nt2) {
      const bf16* vb = sVT + (hh * 32 + nt2 * 16) * AVS;
      v16bf b0 = load_b_frag(vb + 0, AVS);
      v16bf b1 = load_b_frag(vb + 32, AVS);
      v8f av = wmma_bf16(aP0, b0, v8f_zero());
      av = wmma_bf16(aP1, b1, av);
#pragma unroll
      for (int g = 0; g < 8; ++g) {
        int tok = mt * 16 + g + halfadd;
        int col = hh * 32 + nt2 * 16 + nlane;
        AF[qbase + (size_t)tok * DIMC + col] = (bf16)av[g];
      }
    }
    // ---- mf = P @ ce (N = 16 motion channels), minus own ce
    const bf16* cb = sCT + (hh * 16) * AVS;
    v16bf c0 = load_b_frag(cb + 0, AVS);
    v16bf c1 = load_b_frag(cb + 32, AVS);
    v8f ac = wmma_bf16(aP0, c0, v8f_zero());
    ac = wmma_bf16(aP1, c1, ac);
#pragma unroll
    for (int g = 0; g < 8; ++g) {
      int tok = mt * 16 + g + halfadd;
      int col = hh * 16 + nlane;
      float mv = ac[g] - CE[cbase + (size_t)tok * MDIM + col];
      MF[cbase + (size_t)tok * MDIM + col] = (bf16)mv;
    }
  }
}

// ---------------------------------------------------------------------------
// Depthwise 3x3 (SAME) + exact GELU. 1 block = 1 pixel, 4 channels / thread.
// ---------------------------------------------------------------------------
__global__ __launch_bounds__(256) void mfb_dwconv_gelu(
    const bf16* __restrict__ Hb, const float* __restrict__ Wdw,
    const float* __restrict__ bdw, bf16* __restrict__ Gb) {
  int pid = blockIdx.x;
  int b = pid >> 14, rem = pid & 16383;
  int y = rem >> 7, x = rem & 127;
  int c0 = threadIdx.x * 4;
  float s[4];
#pragma unroll
  for (int j = 0; j < 4; ++j) s[j] = bdw[c0 + j];
#pragma unroll
  for (int ky = 0; ky < 3; ++ky) {
    int yy = y + ky - 1;
    if ((unsigned)yy >= 128u) continue;
#pragma unroll
    for (int kx = 0; kx < 3; ++kx) {
      int xx = x + kx - 1;
      if ((unsigned)xx >= 128u) continue;
      const bf16* hp =
          Hb + ((size_t)(b << 14) + (size_t)yy * 128 + xx) * HIDDEN + c0;
      const float* wp = Wdw + (ky * 3 + kx) * HIDDEN + c0;
#pragma unroll
      for (int j = 0; j < 4; ++j) s[j] += (float)hp[j] * wp[j];
    }
  }
#pragma unroll
  for (int j = 0; j < 4; ++j) {
    float u = s[j];
    float ge = 0.5f * u * (1.0f + erff(u * 0.70710678118654752f));
    Gb[(size_t)pid * HIDDEN + c0 + j] = (bf16)ge;
  }
}

// ---------------------------------------------------------------------------
// Host orchestration
// ---------------------------------------------------------------------------
extern "C" void kernel_launch(void* const* d_in, const int* in_sizes, int n_in,
                              void* d_out, int out_size, void* d_ws,
                              size_t ws_size, hipStream_t stream) {
  (void)in_sizes; (void)n_in; (void)out_size; (void)ws_size;
  const float* x     = (const float*)d_in[0];
  const float* cor   = (const float*)d_in[1];
  const float* g1    = (const float*)d_in[2];
  const float* b1    = (const float*)d_in[3];
  const float* Wq    = (const float*)d_in[4];
  const float* bq    = (const float*)d_in[5];
  const float* Wkv   = (const float*)d_in[6];
  const float* bkv   = (const float*)d_in[7];
  const float* Wproj = (const float*)d_in[8];
  const float* bproj = (const float*)d_in[9];
  const float* Wce   = (const float*)d_in[10];
  const float* bce   = (const float*)d_in[11];
  const float* Wmp   = (const float*)d_in[12];
  const float* bmp   = (const float*)d_in[13];
  const float* g2    = (const float*)d_in[14];
  const float* b2    = (const float*)d_in[15];
  const float* Wfc1  = (const float*)d_in[16];
  const float* bfc1  = (const float*)d_in[17];
  const float* Wdw   = (const float*)d_in[18];
  const float* bdw   = (const float*)d_in[19];
  const float* Wfc2  = (const float*)d_in[20];
  const float* bfc2  = (const float*)d_in[21];

  float* outX = (float*)d_out;                 // (B, H*W, 256)
  float* outM = outX + (size_t)NTOK * DIMC;    // (B, H*W, 128)

  char* ws = (char*)d_ws;
  // phase-1 buffers
  float* xn  = (float*)(ws + 0);           // 134.2 MB  x_norm (window rows)
  float* cef = (float*)(ws + 134217728ull);// 67.1 MB   ce (window rows)
  bf16* qb   = (bf16*)(ws + 201326592ull); // 67.1 MB   q bf16
  bf16* kvb  = (bf16*)(ws + 268435456ull); // 134.2 MB  k|v bf16
  bf16* afb  = (bf16*)(ws + 402653184ull); // 67.1 MB   attn@v bf16
  bf16* mbf  = (bf16*)(ws + 469762048ull); // 33.6 MB   attn@ce - ce bf16
  // phase-2 buffers (reuse phase-1 regions; ordering makes this safe)
  bf16* hb   = (bf16*)(ws + 0);            // 268.4 MB  fc1 out
  bf16* gb   = (bf16*)(ws + 268435456ull); // 268.4 MB  dwconv+gelu out
  bf16* xn2  = (bf16*)(ws + 402653184ull); // 67.1 MB   ln2 out (dead before gb write)
  // transposed bf16 weights (persist)
  bf16* wts    = (bf16*)(ws + 536870912ull);
  bf16* WqT    = wts + 0;
  bf16* WkvT   = wts + 65536;
  bf16* WprojT = wts + 196608;
  bf16* WmpT   = wts + 262144;
  bf16* Wfc1T  = wts + 278528;
  bf16* Wfc2T  = wts + 540672;

  // weight prep
  mfb_transpose<<<256, 256, 0, stream>>>(Wq, WqT, 256, 256);
  mfb_transpose<<<512, 256, 0, stream>>>(Wkv, WkvT, 256, 512);
  mfb_transpose<<<256, 256, 0, stream>>>(Wproj, WprojT, 256, 256);
  mfb_transpose<<<64, 256, 0, stream>>>(Wmp, WmpT, 128, 128);
  mfb_transpose<<<1024, 256, 0, stream>>>(Wfc1, Wfc1T, 256, 1024);
  mfb_transpose<<<1024, 256, 0, stream>>>(Wfc2, Wfc2T, 1024, 256);

  const int M = NTOK;
  // 1) LN1 + coordinate embedding (window layout)
  mfb_ln1_ce<<<16384, 256, 0, stream>>>(x, cor, g1, b1, Wce, bce, xn, cef);
  // 2) q / kv projections (bf16 WMMA)
  mfb_gemm<float><<<dim3(2, 1024), 256, 0, stream>>>(
      xn, WqT, bq, nullptr, nullptr, qb, M, 256, 256, 0);
  mfb_gemm<float><<<dim3(4, 1024), 256, 0, stream>>>(
      xn, WkvT, bkv, nullptr, nullptr, kvb, M, 512, 256, 0);
  // 3) windowed attention (appearance + motion streams share one softmax)
  mfb_attn<<<NWIN, 256, 0, stream>>>(qb, kvb, cef, afb, mbf);
  // 4) output proj + residual (x_norm + af), window->flat remap into d_out
  mfb_gemm<bf16><<<dim3(2, 1024), 256, 0, stream>>>(
      afb, WprojT, bproj, xn, outX, nullptr, M, 256, 256, 1);
  // 5) motion proj, remap into d_out (second output)
  mfb_gemm<bf16><<<dim3(1, 1024), 256, 0, stream>>>(
      mbf, WmpT, bmp, nullptr, outM, nullptr, M, 128, 128, 1);
  // 6) MLP: LN2 -> fc1 -> dwconv3x3+gelu -> fc2 (+ residual into d_out)
  mfb_ln2<<<16384, 256, 0, stream>>>(outX, g2, b2, xn2);
  mfb_gemm<bf16><<<dim3(8, 1024), 256, 0, stream>>>(
      xn2, Wfc1T, bfc1, nullptr, nullptr, hb, M, 1024, 256, 0);
  mfb_dwconv_gelu<<<131072, 256, 0, stream>>>(hb, Wdw, bdw, gb);
  mfb_gemm<bf16><<<dim3(2, 1024), 256, 0, stream>>>(
      gb, Wfc2T, bfc2, outX, outX, nullptr, M, 256, 1024, 0);
}
